// QRNNLayer_57750130262421
// MI455X (gfx1250) — compile-verified, matrix operand order checked by reading, not run on hardware
//
#include <hip/hip_runtime.h>
#include <hip/hip_bf16.h>
#include <math.h>

// ---------------------------------------------------------------------------
// QRNN layer for MI455X (gfx1250, wave32, WMMA).
//   Phase 1: fused implicit-GEMM conv (z and f convs concatenated in N):
//            raw[m, n] = sum_k sum_c xpad[b, t+k, c] * W[n, c, k] + bias[n]
//            M = B*T = 32768, N = 2H = 1024, K = 3*C = 1536, f16 in / f32 acc
//            via v_wmma_f32_16x16x32_f16, double-buffered
//            GLOBAL_LOAD_ASYNC_TO_LDS_B128 staging (ASYNCcnt pipeline).
//   Phase 2: log-domain scans. With P_t = cumsum(log_sigmoid(raw_f)) (t>=1),
//            c_t = log(softplus(raw_z)) + log_sigmoid(-raw_f):
//            out_t = exp( logcumsumexp_{s<=t}(c_s - P_s) + P_t )
//            (the reference's suffix-sum rg cancels algebraically).
// ---------------------------------------------------------------------------

typedef _Float16 v16h __attribute__((ext_vector_type(16)));
typedef float    v8f  __attribute__((ext_vector_type(8)));
typedef int      v4i  __attribute__((ext_vector_type(4)));

static constexpr int Bb   = 8;
static constexpr int Tt   = 4096;
static constexpr int Cc   = 512;
static constexpr int Hh   = 512;
static constexpr int Tpad = Tt + 2;
static constexpr int Nn   = 1024;        // [0,512) = z channels, [512,1024) = f
static constexpr int KK   = 1536;        // 3 * C
static constexpr int CHUNK = 256;
static constexpr int NCH   = Tt / CHUNK; // 16
static constexpr int NLANE = Bb * Hh;    // 4096 scan lanes

// workspace offsets (bytes)
static constexpr size_t OFF_XP   = 0;                                   // f16 xpad  (B,Tpad,C)
static constexpr size_t SZ_XP    = (size_t)Bb * Tpad * Cc * 2;          // 33,570,816
static constexpr size_t OFF_W    = OFF_XP + SZ_XP;                      // f16 W^T   (N,KK)
static constexpr size_t SZ_W     = (size_t)Nn * KK * 2;                 //  3,145,728
static constexpr size_t OFF_RAW  = OFF_W + SZ_W;                        // f32 raw   (M,N)
static constexpr size_t SZ_RAW   = (size_t)Bb * Tt * Nn * 4;            // 134,217,728
static constexpr size_t OFF_AGGP = OFF_RAW + SZ_RAW;
static constexpr size_t SZ_AGG   = (size_t)NLANE * NCH * 4;             // 262,144
static constexpr size_t OFF_AGGL = OFF_AGGP + SZ_AGG;

// ---------------------------------------------------------------------------
// stable log helpers
// ---------------------------------------------------------------------------
__device__ __forceinline__ float softplus_f(float x) {
    return fmaxf(x, 0.0f) + log1pf(expf(-fabsf(x)));
}
__device__ __forceinline__ float logsig_f(float x) {      // log(sigmoid(x))
    return -softplus_f(-x);
}
__device__ __forceinline__ float log_softplus_f(float x) {
    if (x < -12.0f) return x;                             // softplus(x) ~ e^x
    return logf(softplus_f(x));
}

// async global -> LDS copy, 16B per lane, tracked by ASYNCcnt
__device__ __forceinline__ void async_copy_b128(const _Float16* g, _Float16* l) {
    __builtin_amdgcn_global_load_async_to_lds_b128(
        (__attribute__((address_space(1))) v4i*)(g),
        (__attribute__((address_space(3))) v4i*)(l), 0, 0);
}

// ---------------------------------------------------------------------------
// pack kernels: build f16 padded input and f16 transposed weights
// ---------------------------------------------------------------------------
__global__ __launch_bounds__(256) void pack_x_kernel(const float* __restrict__ x,
                                                     _Float16* __restrict__ xp) {
    int e = blockIdx.x * 256 + threadIdx.x;
    int total = Bb * Tpad * Cc;
    if (e >= total) return;
    int b = e / (Tpad * Cc);
    int r = e - b * (Tpad * Cc);
    int t = r / Cc;
    int c = r - t * Cc;
    float v = 0.0f;
    if (t > 0 && t < Tpad - 1)
        v = x[((size_t)b * Tt + (t - 1)) * Cc + c];
    xp[e] = (_Float16)v;
}

__global__ __launch_bounds__(256) void pack_w_kernel(const float* __restrict__ zw,
                                                     const float* __restrict__ fw,
                                                     _Float16* __restrict__ wt) {
    int e = blockIdx.x * 256 + threadIdx.x;
    int total = Nn * KK;
    if (e >= total) return;
    int n  = e / KK;
    int kk = e - n * KK;
    int k  = kk >> 9;        // tap 0..2
    int c  = kk & 511;
    float v;
    if (n < Hh) v = zw[((size_t)n * Cc + c) * 3 + k];
    else        v = fw[((size_t)(n - Hh) * Cc + c) * 3 + k];
    wt[e] = (_Float16)v;     // layout: wt[n*KK + k*512 + c]
}

// ---------------------------------------------------------------------------
// GEMM: block tile 128x256, 8 waves (2 M x 4 N), wave tile 64x64,
// K staged 32-wide, double-buffered async LDS staging.
// Key trick: for both A and B the global address of a K-slice is
// (per-thread base) + kk, because kk = tap*512 + c and the tap's t-shift
// contributes exactly tap*512 in the padded input.
// ---------------------------------------------------------------------------
static constexpr int LDR    = 40;                 // halfs per LDS row (80B, 16B-aligned)
static constexpr int A_ROWS = 128;
static constexpr int B_ROWS = 256;
static constexpr int STAGE  = (A_ROWS + B_ROWS) * LDR;   // 15360 halfs = 30KB / buffer

union frag_cvt { uint4 u[2]; v16h h; };

__global__ __launch_bounds__(256) void gemm_kernel(const _Float16* __restrict__ xp,
                                                   const _Float16* __restrict__ wt,
                                                   const float* __restrict__ zb,
                                                   const float* __restrict__ fb,
                                                   float* __restrict__ raw) {
    __shared__ _Float16 lds[2 * STAGE];           // 61,440 B, double buffered

    const int tid   = threadIdx.x;
    const int lane  = tid & 31;
    const int wave  = tid >> 5;
    const int waveM = wave & 1;          // 2 wave-rows of 64
    const int waveN = wave >> 1;         // 4 wave-cols of 64
    const int l16   = lane & 15;
    const int hi16  = lane >> 4;         // 0 or 1

    const int mbase = blockIdx.x * 128;  // M block (T multiple of 128 -> one batch)
    const int nbase = blockIdx.y * 256;  // N block
    const int b     = mbase / Tt;
    const int t0    = mbase % Tt;

    // per-thread async-staging bases: stage address = base + kk
    // A: 128 rows x 32 halfs = 512 segs of 8 halfs -> 2 segs/thread
    // B: 256 rows x 32 halfs = 1024 segs           -> 4 segs/thread
    const _Float16* gA[2];
    const _Float16* gB[4];
    int lA[2], lB[4];
    #pragma unroll
    for (int s = 0; s < 2; ++s) {
        int idx = tid + s * 256;
        int r = idx >> 2, sc = idx & 3;
        gA[s] = xp + ((size_t)(b * Tpad + t0 + r) * Cc) + sc * 8;
        lA[s] = r * LDR + sc * 8;
    }
    #pragma unroll
    for (int s = 0; s < 4; ++s) {
        int idx = tid + s * 256;
        int r = idx >> 2, sc = idx & 3;
        gB[s] = wt + ((size_t)(nbase + r) * KK) + sc * 8;
        lB[s] = A_ROWS * LDR + r * LDR + sc * 8;
    }

    v8f acc[4][4];
    #pragma unroll
    for (int mi = 0; mi < 4; ++mi)
        #pragma unroll
        for (int ni = 0; ni < 4; ++ni)
            acc[mi][ni] = (v8f){0.f, 0.f, 0.f, 0.f, 0.f, 0.f, 0.f, 0.f};

    constexpr int NIT = KK / 32;         // 48 K-slices
    // prologue: stage 0 into buffer 0
    {
        #pragma unroll
        for (int s = 0; s < 2; ++s) async_copy_b128(gA[s], &lds[lA[s]]);
        #pragma unroll
        for (int s = 0; s < 4; ++s) async_copy_b128(gB[s], &lds[lB[s]]);
    }

    for (int it = 0; it < NIT; ++it) {
        const int cur = it & 1;
        if (it + 1 < NIT) {
            const int kk  = (it + 1) * 32;
            const int off = (cur ^ 1) * STAGE;
            #pragma unroll
            for (int s = 0; s < 2; ++s) async_copy_b128(gA[s] + kk, &lds[off + lA[s]]);
            #pragma unroll
            for (int s = 0; s < 4; ++s) async_copy_b128(gB[s] + kk, &lds[off + lB[s]]);
            __builtin_amdgcn_s_wait_asynccnt(6);   // current stage landed (in-order)
        } else {
            __builtin_amdgcn_s_wait_asynccnt(0);
        }
        __syncthreads();

        const _Float16* Ab = &lds[cur * STAGE];
        const _Float16* Bbuf = Ab + A_ROWS * LDR;
        const int koff = hi16 * 8;       // lanes<16: K{0..7,16..23}; lanes>=16: K{8..15,24..31}
        v16h af[4], bf[4];
        #pragma unroll
        for (int mi = 0; mi < 4; ++mi) {
            int row = waveM * 64 + mi * 16 + l16;
            frag_cvt cv;
            cv.u[0] = *reinterpret_cast<const uint4*>(&Ab[row * LDR + koff]);
            cv.u[1] = *reinterpret_cast<const uint4*>(&Ab[row * LDR + 16 + koff]);
            af[mi] = cv.h;
        }
        #pragma unroll
        for (int ni = 0; ni < 4; ++ni) {
            int row = waveN * 64 + ni * 16 + l16;
            frag_cvt cv;
            cv.u[0] = *reinterpret_cast<const uint4*>(&Bbuf[row * LDR + koff]);
            cv.u[1] = *reinterpret_cast<const uint4*>(&Bbuf[row * LDR + 16 + koff]);
            bf[ni] = cv.h;
        }
        #pragma unroll
        for (int mi = 0; mi < 4; ++mi)
            #pragma unroll
            for (int ni = 0; ni < 4; ++ni)
                acc[mi][ni] = __builtin_amdgcn_wmma_f32_16x16x32_f16(
                    false, af[mi], false, bf[ni],
                    (short)0, acc[mi][ni], false, false);
        __syncthreads();
    }

    // epilogue: add bias, store raw (lanes 0-15 -> M=r, lanes 16-31 -> M=8+r)
    int   nglob[4];
    float bias[4];
    #pragma unroll
    for (int ni = 0; ni < 4; ++ni) {
        int n = nbase + waveN * 64 + ni * 16 + l16;
        nglob[ni] = n;
        bias[ni]  = (n < Hh) ? zb[n] : fb[n - Hh];
    }
    #pragma unroll
    for (int mi = 0; mi < 4; ++mi) {
        #pragma unroll
        for (int ni = 0; ni < 4; ++ni) {
            #pragma unroll
            for (int r = 0; r < 8; ++r) {
                int mg = mbase + waveM * 64 + mi * 16 + (hi16 << 3) + r;
                raw[(size_t)mg * Nn + nglob[ni]] = acc[mi][ni][r] + bias[ni];
            }
        }
    }
}

// ---------------------------------------------------------------------------
// scan pass 1: per (lane, chunk) -> aggregates (dP, logsumexp(c - P_local))
// ---------------------------------------------------------------------------
__global__ __launch_bounds__(256) void scan1_kernel(const float* __restrict__ raw,
                                                    float* __restrict__ aggP,
                                                    float* __restrict__ aggL) {
    int gid = blockIdx.x * 256 + threadIdx.x;          // 0 .. 65535
    int chunk   = gid >> 12;
    int laneIdx = gid & 4095;
    int b = laneIdx >> 9;
    int h = laneIdx & 511;
    int t0 = chunk * CHUNK;

    const float* rp = raw + ((size_t)b * Tt + t0) * Nn + h;
    float P = 0.f, m = -INFINITY, s = 0.f;
    for (int i = 0; i < CHUNK; ++i) {
        float rz = rp[0];
        float rf = rp[Hh];
        int t = t0 + i;
        float a = (t == 0) ? 0.f : logsig_f(rf);
        P += a;
        float cv = log_softplus_f(rz) + logsig_f(-rf);
        float v  = cv - P;
        if (v > m) { s = s * expf(m - v) + 1.f; m = v; }
        else       { s += expf(v - m); }
        rp += Nn;
    }
    aggP[chunk * NLANE + laneIdx] = P;
    aggL[chunk * NLANE + laneIdx] = m + logf(s);
}

// ---------------------------------------------------------------------------
// middle scan: sequential over 16 chunk aggregates per lane; replaces
// aggregates with per-chunk carry-in (P0, Lin).
// ---------------------------------------------------------------------------
__global__ __launch_bounds__(256) void scanmid_kernel(float* __restrict__ aggP,
                                                      float* __restrict__ aggL) {
    int laneIdx = blockIdx.x * 256 + threadIdx.x;      // 0 .. 4095
    float cP = 0.f, cL = -INFINITY;
    for (int ch = 0; ch < NCH; ++ch) {
        int idx = ch * NLANE + laneIdx;
        float dP = aggP[idx];
        float dL = aggL[idx];
        aggP[idx] = cP;                                // carry-in for chunk ch
        aggL[idx] = cL;
        float x1 = cL, x2 = dL - cP;                   // logaddexp
        float mm = fmaxf(x1, x2);
        if (mm > -1e30f)
            cL = mm + logf(expf(x1 - mm) + expf(x2 - mm));
        cP += dP;
    }
}

// ---------------------------------------------------------------------------
// scan pass 2: redo local scan with carry-in, emit out = exp(L_t + P_t)
// out = exp(Lin + P0 + Pl) + s * exp(m + Pl)
// ---------------------------------------------------------------------------
__global__ __launch_bounds__(256) void scan2_kernel(const float* __restrict__ raw,
                                                    const float* __restrict__ aggP,
                                                    const float* __restrict__ aggL,
                                                    float* __restrict__ out) {
    int gid = blockIdx.x * 256 + threadIdx.x;
    int chunk   = gid >> 12;
    int laneIdx = gid & 4095;
    int b = laneIdx >> 9;
    int h = laneIdx & 511;
    int t0 = chunk * CHUNK;

    float P0  = aggP[chunk * NLANE + laneIdx];
    float Lin = aggL[chunk * NLANE + laneIdx];
    bool  hasIn = (Lin > -1e30f);

    const float* rp = raw + ((size_t)b * Tt + t0) * Nn + h;
    float* op = out + ((size_t)b * Tt + t0) * Hh + h;
    float Pl = 0.f, m = -INFINITY, s = 0.f;
    for (int i = 0; i < CHUNK; ++i) {
        float rz = rp[0];
        float rf = rp[Hh];
        int t = t0 + i;
        float a = (t == 0) ? 0.f : logsig_f(rf);
        Pl += a;
        float cv = log_softplus_f(rz) + logsig_f(-rf);
        float v  = cv - Pl;
        if (v > m) { s = s * expf(m - v) + 1.f; m = v; }
        else       { s += expf(v - m); }
        float o = s * expf(m + Pl);
        if (hasIn) o += expf(Lin + P0 + Pl);
        *op = o;
        rp += Nn;
        op += Hh;
    }
}

// ---------------------------------------------------------------------------
// launcher
// ---------------------------------------------------------------------------
extern "C" void kernel_launch(void* const* d_in, const int* in_sizes, int n_in,
                              void* d_out, int out_size, void* d_ws, size_t ws_size,
                              hipStream_t stream) {
    const float* x  = (const float*)d_in[0];   // (B,T,C)
    const float* zw = (const float*)d_in[1];   // (H,C,3)
    const float* zb = (const float*)d_in[2];   // (H)
    const float* fw = (const float*)d_in[3];   // (H,C,3)
    const float* fb = (const float*)d_in[4];   // (H)
    float* out = (float*)d_out;                // (B,T,H)

    char* ws = (char*)d_ws;
    _Float16* xp   = (_Float16*)(ws + OFF_XP);
    _Float16* wt   = (_Float16*)(ws + OFF_W);
    float*    raw  = (float*)(ws + OFF_RAW);
    float*    aggP = (float*)(ws + OFF_AGGP);
    float*    aggL = (float*)(ws + OFF_AGGL);

    {   // pack padded f16 input
        int total = Bb * Tpad * Cc;
        pack_x_kernel<<<(total + 255) / 256, 256, 0, stream>>>(x, xp);
    }
    {   // pack transposed f16 weights (z|f concatenated in N)
        int total = Nn * KK;
        pack_w_kernel<<<(total + 255) / 256, 256, 0, stream>>>(zw, fw, wt);
    }
    {   // implicit-GEMM conv: grid 256 (M) x 4 (N)
        dim3 grid(Bb * Tt / 128, Nn / 256);
        gemm_kernel<<<grid, 256, 0, stream>>>(xp, wt, zb, fb, raw);
    }
    scan1_kernel<<<NLANE * NCH / 256, 256, 0, stream>>>(raw, aggP, aggL);
    scanmid_kernel<<<NLANE / 256, 256, 0, stream>>>(aggP, aggL);
    scan2_kernel<<<NLANE * NCH / 256, 256, 0, stream>>>(raw, aggP, aggL, out);
}
